// DecisionTransformerBlock_24189255811129
// MI455X (gfx1250) — matrix operand order repaired and checked by reading, not compiled, not run on hardware
//
#include <hip/hip_runtime.h>
#include <hip/hip_bf16.h>

typedef __attribute__((ext_vector_type(16))) _Float16 v16h;
typedef __attribute__((ext_vector_type(8)))  _Float16 v8h;
typedef __attribute__((ext_vector_type(8)))  float    v8f;
typedef __attribute__((ext_vector_type(4)))  int      i32x4;

#define LDST 40                 // LDS row stride in halves (80B -> 16B aligned)
#define TILEH (128 * LDST)      // halves per 128x32 staged tile
#define NEG_INF (-__builtin_huge_valf())

#if __has_builtin(__builtin_amdgcn_global_load_async_to_lds_b128)
#define HAS_ASYNC_LDS 1
#else
#define HAS_ASYNC_LDS 0
#endif

static constexpr int Bn = 4, Nn = 2048, Cc = 1024, Hh = 16, Dd = 64, FFn = 4096;
static constexpr int Mrows = Bn * Nn;   // 8192

__device__ __forceinline__ v16h make_v16h(v8h lo, v8h hi) {
  union { v16h v; v8h h[2]; } u; u.h[0] = lo; u.h[1] = hi; return u.v;
}
__device__ __forceinline__ v8f wmma_f16(v16h a, v16h b, v8f c) {
  return __builtin_amdgcn_wmma_f32_16x16x32_f16(false, a, false, b, (short)0, c, false, false);
}
__device__ __forceinline__ float redmax16(float v) {
#pragma unroll
  for (int off = 1; off < 16; off <<= 1) v = fmaxf(v, __shfl_xor(v, off, 32));
  return v;
}
__device__ __forceinline__ float redsum16(float v) {
#pragma unroll
  for (int off = 1; off < 16; off <<= 1) v += __shfl_xor(v, off, 32);
  return v;
}

// ---- tile staging: 256 threads move a 128x32 half tile (K contiguous) into LDS ----
__device__ __forceinline__ void stage_tile(const _Float16* __restrict__ src, int ldK,
                                           int row0, int k0, _Float16* dst) {
  int t = threadIdx.x;
#pragma unroll
  for (int i = 0; i < 2; ++i) {
    int idx = t + i * 256;            // 0..511
    int r = idx >> 2, cc = idx & 3;   // row, 8-half chunk
    const _Float16* g = src + (size_t)(row0 + r) * ldK + k0 + cc * 8;
    _Float16* l = dst + r * LDST + cc * 8;
#if HAS_ASYNC_LDS
    // memory -> LDS directly via the CDNA5 async path (tracked by ASYNCcnt)
    __builtin_amdgcn_global_load_async_to_lds_b128(
        (__attribute__((address_space(1))) i32x4*)g,
        (__attribute__((address_space(3))) i32x4*)l, 0, 0);
#else
    *(v8h*)l = *(const v8h*)g;
#endif
  }
}
__device__ __forceinline__ void wait_async() {
#if HAS_ASYNC_LDS
  asm volatile("s_wait_asynccnt 0" ::: "memory");
#endif
}

// A fragment (16x32 f16): lane m holds K = hi*16..hi*16+15
__device__ __forceinline__ v16h load_a_frag(const _Float16* As, int mlocal, int hi) {
  v8h lo = *(const v8h*)(As + mlocal * LDST + hi * 16);
  v8h hh = *(const v8h*)(As + mlocal * LDST + hi * 16 + 8);
  return make_v16h(lo, hh);
}
// B fragment (32x16): lane n holds K = hi*16..hi*16+15 (NT weight row = contiguous)
__device__ __forceinline__ v16h load_b_frag(const _Float16* Bs, int nlocal, int hi) {
  v8h lo = *(const v8h*)(Bs + nlocal * LDST + hi * 16);
  v8h hh = *(const v8h*)(Bs + nlocal * LDST + hi * 16 + 8);
  return make_v16h(lo, hh);
}

// 128x128 block tile, 8 waves (2 M-rows x 4 N-cols), wave tile 64x32 = 4x2 WMMAs.
// Double-buffered: async tile k+1 streams in while WMMAs consume tile k.
__device__ __forceinline__ void gemm_core(const _Float16* __restrict__ A,
                                          const _Float16* __restrict__ W, int K,
                                          int blockM, int blockN,
                                          _Float16* As, _Float16* Bs, v8f acc[4][2]) {
  int lane = threadIdx.x & 31, w = threadIdx.x >> 5;
  int waveM = w >> 2, waveN = w & 3;
  int m15 = lane & 15, hi = lane >> 4;
#pragma unroll
  for (int i = 0; i < 4; ++i)
#pragma unroll
    for (int j = 0; j < 2; ++j)
#pragma unroll
      for (int e = 0; e < 8; ++e) acc[i][j][e] = 0.f;

  stage_tile(A, K, blockM, 0, As);
  stage_tile(W, K, blockN, 0, Bs);
  int nk = K >> 5;
  for (int kb = 0; kb < nk; ++kb) {
    wait_async();
    __syncthreads();                      // tile kb resident; all waves past compute kb-1
    int cur = kb & 1, nxt = cur ^ 1;
    if (kb + 1 < nk) {
      stage_tile(A, K, blockM, (kb + 1) << 5, As + nxt * TILEH);
      stage_tile(W, K, blockN, (kb + 1) << 5, Bs + nxt * TILEH);
    }
    const _Float16* Ac = As + cur * TILEH;
    const _Float16* Bc = Bs + cur * TILEH;
    v16h bf[2];
#pragma unroll
    for (int j = 0; j < 2; ++j) bf[j] = load_b_frag(Bc, waveN * 32 + j * 16 + m15, hi);
#pragma unroll
    for (int i = 0; i < 4; ++i) {
      v16h af = load_a_frag(Ac, waveM * 64 + i * 16 + m15, hi);
#pragma unroll
      for (int j = 0; j < 2; ++j) acc[i][j] = wmma_f16(af, bf[j], acc[i][j]);
    }
  }
}

// ---------------- elementwise / LN ----------------
__global__ void k_f32_to_f16(const float* __restrict__ s, _Float16* __restrict__ d, int n) {
  int i = blockIdx.x * blockDim.x + threadIdx.x;
  if (i < n) d[i] = (_Float16)s[i];
}

__global__ __launch_bounds__(256) void k_ln(const float* __restrict__ X,
                                            const float* __restrict__ g,
                                            const float* __restrict__ b,
                                            _Float16* __restrict__ out) {
  __shared__ float ss[8], ss2[8];
  int row = blockIdx.x;
  int lane = threadIdx.x & 31, w = threadIdx.x >> 5;
  const float* x = X + (size_t)row * Cc;
  float v[4], s = 0.f, s2 = 0.f;
#pragma unroll
  for (int i = 0; i < 4; ++i) {
    v[i] = x[threadIdx.x + i * 256];
    s += v[i]; s2 += v[i] * v[i];
  }
#pragma unroll
  for (int off = 1; off < 32; off <<= 1) { s += __shfl_xor(s, off, 32); s2 += __shfl_xor(s2, off, 32); }
  if (lane == 0) { ss[w] = s; ss2[w] = s2; }
  __syncthreads();
  float ts = 0.f, ts2 = 0.f;
#pragma unroll
  for (int i = 0; i < 8; ++i) { ts += ss[i]; ts2 += ss2[i]; }
  float mu = ts * (1.f / Cc);
  float var = ts2 * (1.f / Cc) - mu * mu;
  float rstd = rsqrtf(var + 1e-6f);
#pragma unroll
  for (int i = 0; i < 4; ++i) {
    int c = threadIdx.x + i * 256;
    out[(size_t)row * Cc + c] = (_Float16)((v[i] - mu) * rstd * g[c] + b[c]);
  }
}

// ---------------- QKV GEMM: scatter q/k (BHND) and v transposed (BHDN) ----------------
__global__ __launch_bounds__(256) void k_gemm_qkv(const _Float16* __restrict__ A,
                                                  const _Float16* __restrict__ W,
                                                  const float* __restrict__ bias,
                                                  _Float16* __restrict__ q16,
                                                  _Float16* __restrict__ k16,
                                                  _Float16* __restrict__ vT) {
  __shared__ _Float16 As[2 * TILEH];
  __shared__ _Float16 Bs[2 * TILEH];
  int blockM = blockIdx.y * 128, blockN = blockIdx.x * 128;
  v8f acc[4][2];
  gemm_core(A, W, Cc, blockM, blockN, As, Bs, acc);
  int lane = threadIdx.x & 31, w = threadIdx.x >> 5;
  int waveM = w >> 2, waveN = w & 3;
  int m15 = lane & 15, hi = lane >> 4;
#pragma unroll
  for (int i = 0; i < 4; ++i)
#pragma unroll
    for (int j = 0; j < 2; ++j)
#pragma unroll
      for (int r = 0; r < 8; ++r) {
        int gM = blockM + waveM * 64 + i * 16 + 8 * hi + r;
        int gN = blockN + waveN * 32 + j * 16 + m15;
        float val = acc[i][j][r] + bias[gN];
        int bb = gM >> 11, nseq = gM & 2047;
        int which = gN >> 10, c = gN & 1023, hh = c >> 6, dd = c & 63;
        if (which == 0)
          q16[(((size_t)(bb * Hh + hh)) * Nn + nseq) * Dd + dd] = (_Float16)(val * 0.125f);
        else if (which == 1)
          k16[(((size_t)(bb * Hh + hh)) * Nn + nseq) * Dd + dd] = (_Float16)val;
        else
          vT[(((size_t)(bb * Hh + hh)) * Dd + dd) * Nn + nseq] = (_Float16)val;
      }
}

// ---------------- flash attention: 4 waves x 16 queries, 32 keys / iteration ----------------
__global__ __launch_bounds__(128) void k_attn(const _Float16* __restrict__ Q,
                                              const _Float16* __restrict__ Km,
                                              const _Float16* __restrict__ Vt,
                                              _Float16* __restrict__ O) {
  __shared__ _Float16 pbuf[4 * 16 * LDST];
  int lane = threadIdx.x & 31, w = threadIdx.x >> 5;
  int m15 = lane & 15, hi = lane >> 4;
  int bh = blockIdx.y;
  int qbase = blockIdx.x * 64 + w * 16;
  const _Float16* Qh = Q + (size_t)bh * Nn * Dd;
  const _Float16* Kh = Km + (size_t)bh * Nn * Dd;
  const _Float16* Vh = Vt + (size_t)bh * Dd * Nn;
  _Float16* pw = pbuf + w * 16 * LDST;

  v16h qa[2];
#pragma unroll
  for (int kc = 0; kc < 2; ++kc) {
    const _Float16* p = Qh + (size_t)(qbase + m15) * Dd + kc * 32 + hi * 16;
    qa[kc] = make_v16h(*(const v8h*)p, *(const v8h*)(p + 8));
  }
  float mrow[8], lrow[8];
  v8f oacc[4];
#pragma unroll
  for (int r = 0; r < 8; ++r) { mrow[r] = NEG_INF; lrow[r] = 0.f; }
#pragma unroll
  for (int t = 0; t < 4; ++t)
#pragma unroll
    for (int e = 0; e < 8; ++e) oacc[t][e] = 0.f;

  int kbmax = (qbase + 15) >> 5;
  for (int kb = 0; kb <= kbmax; ++kb) {
    int kstart = kb << 5;
    v8f s[2];
#pragma unroll
    for (int j = 0; j < 2; ++j) {
#pragma unroll
      for (int e = 0; e < 8; ++e) s[j][e] = 0.f;
#pragma unroll
      for (int kc = 0; kc < 2; ++kc) {
        const _Float16* kp = Kh + (size_t)(kstart + j * 16 + m15) * Dd + kc * 32 + hi * 16;
        v16h bf = make_v16h(*(const v8h*)kp, *(const v8h*)(kp + 8));
        s[j] = wmma_f16(qa[kc], bf, s[j]);
      }
    }
#pragma unroll
    for (int r = 0; r < 8; ++r) {
      int row = qbase + r + 8 * hi;
#pragma unroll
      for (int j = 0; j < 2; ++j) {
        int col = kstart + j * 16 + m15;
        if (col > row) s[j][r] = NEG_INF;
      }
      float t = fmaxf(s[0][r], s[1][r]);
      t = redmax16(t);
      float mnew = fmaxf(mrow[r], t);
      float sc = __expf(mrow[r] - mnew);
      float p0 = __expf(s[0][r] - mnew);
      float p1 = __expf(s[1][r] - mnew);
      float rs = redsum16(p0 + p1);
      lrow[r] = lrow[r] * sc + rs;
      mrow[r] = mnew;
#pragma unroll
      for (int t2 = 0; t2 < 4; ++t2) oacc[t2][r] *= sc;
      pw[(r + 8 * hi) * LDST + m15] = (_Float16)p0;
      pw[(r + 8 * hi) * LDST + 16 + m15] = (_Float16)p1;
    }
    // re-swizzle P (C-layout) -> A fragment through wave-private LDS
    v16h pa = make_v16h(*(const v8h*)(pw + m15 * LDST + hi * 16),
                        *(const v8h*)(pw + m15 * LDST + hi * 16 + 8));
#pragma unroll
    for (int t2 = 0; t2 < 4; ++t2) {
      const _Float16* vp = Vh + (size_t)(t2 * 16 + m15) * Nn + kstart + hi * 16;
      v16h bf = make_v16h(*(const v8h*)vp, *(const v8h*)(vp + 8));
      oacc[t2] = wmma_f16(pa, bf, oacc[t2]);
    }
  }
  int b = bh >> 4, h = bh & 15;
#pragma unroll
  for (int r = 0; r < 8; ++r) {
    float inv = 1.0f / lrow[r];
    int row = qbase + r + 8 * hi;
#pragma unroll
    for (int t2 = 0; t2 < 4; ++t2)
      O[(size_t)(b * Nn + row) * Cc + h * Dd + t2 * 16 + m15] = (_Float16)(oacc[t2][r] * inv);
  }
}

// ---------------- proj GEMM + bias + residual (fp32 out) ----------------
__global__ __launch_bounds__(256) void k_gemm_proj(const _Float16* __restrict__ A,
                                                   const _Float16* __restrict__ W,
                                                   const float* __restrict__ bias,
                                                   const float* __restrict__ X,
                                                   float* __restrict__ out) {
  __shared__ _Float16 As[2 * TILEH];
  __shared__ _Float16 Bs[2 * TILEH];
  int blockM = blockIdx.y * 128, blockN = blockIdx.x * 128;
  v8f acc[4][2];
  gemm_core(A, W, Cc, blockM, blockN, As, Bs, acc);
  int lane = threadIdx.x & 31, w = threadIdx.x >> 5;
  int waveM = w >> 2, waveN = w & 3;
  int m15 = lane & 15, hi = lane >> 4;
#pragma unroll
  for (int i = 0; i < 4; ++i)
#pragma unroll
    for (int j = 0; j < 2; ++j)
#pragma unroll
      for (int r = 0; r < 8; ++r) {
        int gM = blockM + waveM * 64 + i * 16 + 8 * hi + r;
        int gN = blockN + waveN * 32 + j * 16 + m15;
        size_t idx = (size_t)gM * Cc + gN;
        out[idx] = X[idx] + acc[i][j][r] + bias[gN];
      }
}

// ---------------- dual-B GEMM: silu(h@w1^T) * (h@w3^T) -> f16, double-buffered ----------------
__global__ __launch_bounds__(256) void k_gemm_swiglu(const _Float16* __restrict__ A,
                                                     const _Float16* __restrict__ W1,
                                                     const _Float16* __restrict__ W3,
                                                     _Float16* __restrict__ out) {
  __shared__ _Float16 As[2 * TILEH];
  __shared__ _Float16 Bs1[2 * TILEH];
  __shared__ _Float16 Bs3[2 * TILEH];
  int blockM = blockIdx.y * 128, blockN = blockIdx.x * 128;
  int lane = threadIdx.x & 31, w = threadIdx.x >> 5;
  int waveM = w >> 2, waveN = w & 3;
  int m15 = lane & 15, hi = lane >> 4;
  v8f a1[4][2], a3[4][2];
#pragma unroll
  for (int i = 0; i < 4; ++i)
#pragma unroll
    for (int j = 0; j < 2; ++j)
#pragma unroll
      for (int e = 0; e < 8; ++e) { a1[i][j][e] = 0.f; a3[i][j][e] = 0.f; }

  stage_tile(A, Cc, blockM, 0, As);
  stage_tile(W1, Cc, blockN, 0, Bs1);
  stage_tile(W3, Cc, blockN, 0, Bs3);
  int nk = Cc >> 5;
  for (int kb = 0; kb < nk; ++kb) {
    wait_async();
    __syncthreads();
    int cur = kb & 1, nxt = cur ^ 1;
    if (kb + 1 < nk) {
      stage_tile(A, Cc, blockM, (kb + 1) << 5, As + nxt * TILEH);
      stage_tile(W1, Cc, blockN, (kb + 1) << 5, Bs1 + nxt * TILEH);
      stage_tile(W3, Cc, blockN, (kb + 1) << 5, Bs3 + nxt * TILEH);
    }
    const _Float16* Ac = As + cur * TILEH;
    v16h b1[2], b3[2];
#pragma unroll
    for (int j = 0; j < 2; ++j) {
      b1[j] = load_b_frag(Bs1 + cur * TILEH, waveN * 32 + j * 16 + m15, hi);
      b3[j] = load_b_frag(Bs3 + cur * TILEH, waveN * 32 + j * 16 + m15, hi);
    }
#pragma unroll
    for (int i = 0; i < 4; ++i) {
      v16h af = load_a_frag(Ac, waveM * 64 + i * 16 + m15, hi);
#pragma unroll
      for (int j = 0; j < 2; ++j) {
        a1[i][j] = wmma_f16(af, b1[j], a1[i][j]);
        a3[i][j] = wmma_f16(af, b3[j], a3[i][j]);
      }
    }
  }
#pragma unroll
  for (int i = 0; i < 4; ++i)
#pragma unroll
    for (int j = 0; j < 2; ++j)
#pragma unroll
      for (int r = 0; r < 8; ++r) {
        int gM = blockM + waveM * 64 + i * 16 + 8 * hi + r;
        int gN = blockN + waveN * 32 + j * 16 + m15;
        float u = a1[i][j][r], gv = a3[i][j][r];
        float sil = u * (1.0f / (1.0f + __expf(-u)));
        out[(size_t)gM * FFn + gN] = (_Float16)(sil * gv);
      }
}

// ---------------- w2 GEMM + residual accumulate into d_out ----------------
__global__ __launch_bounds__(256) void k_gemm_w2(const _Float16* __restrict__ A,
                                                 const _Float16* __restrict__ W,
                                                 float* __restrict__ out) {
  __shared__ _Float16 As[2 * TILEH];
  __shared__ _Float16 Bs[2 * TILEH];
  int blockM = blockIdx.y * 128, blockN = blockIdx.x * 128;
  v8f acc[4][2];
  gemm_core(A, W, FFn, blockM, blockN, As, Bs, acc);
  int lane = threadIdx.x & 31, w = threadIdx.x >> 5;
  int waveM = w >> 2, waveN = w & 3;
  int m15 = lane & 15, hi = lane >> 4;
#pragma unroll
  for (int i = 0; i < 4; ++i)
#pragma unroll
    for (int j = 0; j < 2; ++j)
#pragma unroll
      for (int r = 0; r < 8; ++r) {
        int gM = blockM + waveM * 64 + i * 16 + 8 * hi + r;
        int gN = blockN + waveN * 32 + j * 16 + m15;
        size_t idx = (size_t)gM * Cc + gN;
        out[idx] += acc[i][j][r];
      }
}

extern "C" void kernel_launch(void* const* d_in, const int* in_sizes, int n_in,
                              void* d_out, int out_size, void* d_ws, size_t ws_size,
                              hipStream_t stream) {
  const float* x      = (const float*)d_in[0];
  // d_in[1] = mask (causal tril) — applied analytically in k_attn
  const float* qkv_w  = (const float*)d_in[2];
  const float* qkv_b  = (const float*)d_in[3];
  const float* proj_w = (const float*)d_in[4];
  const float* proj_b = (const float*)d_in[5];
  const float* ln1_g  = (const float*)d_in[6];
  const float* ln1_b  = (const float*)d_in[7];
  const float* ln2_g  = (const float*)d_in[8];
  const float* ln2_b  = (const float*)d_in[9];
  const float* w1     = (const float*)d_in[10];
  const float* w2     = (const float*)d_in[11];
  const float* w3     = (const float*)d_in[12];
  float* out = (float*)d_out;
  char* ws = (char*)d_ws;

  // workspace layout (bytes)
  _Float16* wqkv16 = (_Float16*)(ws + 0);                       // 3C*C   = 6 MB
  _Float16* wprj16 = (_Float16*)(ws + 6291456);                 // C*C    = 2 MB
  _Float16* w1_16  = (_Float16*)(ws + 8388608);                 // FF*C   = 8 MB
  _Float16* w3_16  = (_Float16*)(ws + 16777216);                // 8 MB
  _Float16* w2_16  = (_Float16*)(ws + 25165824);                // 8 MB
  _Float16* h16    = (_Float16*)(ws + 33554432);                // 16 MB (LN1, reused for LN2)
  _Float16* q16    = (_Float16*)(ws + 50331648);                // 16 MB
  _Float16* k16    = (_Float16*)(ws + 67108864);                // 16 MB
  _Float16* vT16   = (_Float16*)(ws + 83886080);                // 16 MB
  _Float16* o16    = (_Float16*)(ws + 100663296);               // 16 MB
  _Float16* m16    = (_Float16*)(ws + 50331648);                // 64 MB, reuses q/k/vT/o

  auto cvt = [&](const float* s, _Float16* d, int n) {
    k_f32_to_f16<<<(n + 255) / 256, 256, 0, stream>>>(s, d, n);
  };
  cvt(qkv_w, wqkv16, 3 * Cc * Cc);
  cvt(proj_w, wprj16, Cc * Cc);
  cvt(w1, w1_16, FFn * Cc);
  cvt(w3, w3_16, FFn * Cc);
  cvt(w2, w2_16, Cc * FFn);

  // attention sub-block
  k_ln<<<Mrows, 256, 0, stream>>>(x, ln1_g, ln1_b, h16);
  k_gemm_qkv<<<dim3(3 * Cc / 128, Mrows / 128), 256, 0, stream>>>(h16, wqkv16, qkv_b, q16, k16, vT16);
  k_attn<<<dim3(Nn / 64, Bn * Hh), 128, 0, stream>>>(q16, k16, vT16, o16);
  k_gemm_proj<<<dim3(Cc / 128, Mrows / 128), 256, 0, stream>>>(o16, wprj16, proj_b, x, out);

  // SwiGLU MLP sub-block
  k_ln<<<Mrows, 256, 0, stream>>>(out, ln2_g, ln2_b, h16);
  k_gemm_swiglu<<<dim3(FFn / 128, Mrows / 128), 256, 0, stream>>>(h16, w1_16, w3_16, m16);
  k_gemm_w2<<<dim3(Cc / 128, Mrows / 128), 256, 0, stream>>>(m16, w2_16, out);
}
